// CausalSelfAttention_39135742001435
// MI455X (gfx1250) — matrix operand order repaired and checked by reading, not compiled, not run on hardware
//
#include <hip/hip_runtime.h>
#include <hip/hip_bf16.h>

// ---------------------------------------------------------------------------
// CausalSelfAttention on gfx1250 (MI455X): f16 WMMA for all matmul stages,
// flash-attention with double-buffered LDS K/V tiles (K via async Tensor Data
// Mover), copy-free software-pipelined WMMA GEMM. wave32 only.
// ---------------------------------------------------------------------------

typedef __attribute__((ext_vector_type(16))) _Float16     v16h;
typedef __attribute__((ext_vector_type(8)))  _Float16     v8h;
typedef __attribute__((ext_vector_type(8)))  float        v8f;
typedef __attribute__((ext_vector_type(4)))  unsigned int v4u;
typedef __attribute__((ext_vector_type(8)))  int          v8i_;
typedef __attribute__((ext_vector_type(4)))  int          v4i_;

constexpr int Bb = 2;
constexpr int Lq = 2048;
constexpr int Dm = 1024;
constexpr int Hh = 16;
constexpr int HD = 64;              // Dm / Hh
constexpr int BL = Bb * Lq;         // 4096 rows
constexpr int D3 = 3 * Dm;          // 3072

#if defined(__HIP_DEVICE_COMPILE__) && __has_builtin(__builtin_amdgcn_tensor_load_to_lds)
#define USE_TDM 1
#endif

// ---------------------------------------------------------------------------
// WMMA fragment loaders (contiguous k order: e -> k = 16*(lane/16) + e, wave32)
// ---------------------------------------------------------------------------

// A fragment 16x32 (f16), row-major source: lane (l&15) = row, elements:
// e in 0..15 -> K = kb8 + e (kb8 = 16 for lanes 16..31)
__device__ inline v16h frag_a_rowmajor(const _Float16* __restrict__ base, size_t ld,
                                       size_t row0, int col0, int lane) {
  const int kb8 = (lane & 16) ? 16 : 0;
  const _Float16* p = base + (row0 + (size_t)(lane & 15)) * ld + col0 + kb8;
  union { v16h v; v8h h[2]; } u;
  u.h[0] = *(const v8h*)(p);
  u.h[1] = *(const v8h*)(p + 8);
  return u.v;
}

// B fragment 32x16 (f16) where B[k][n] = base[(n_row0+n)*ld + col0 + k].
__device__ inline v16h frag_b_colmajor(const _Float16* __restrict__ base, size_t ld,
                                       size_t n_row0, int col0, int lane) {
  const int kb = (lane & 16) ? 16 : 0;
  const _Float16* p = base + (n_row0 + (size_t)(lane & 15)) * ld + col0 + kb;
  union { v16h v; v8h h[2]; } u;
  u.h[0] = *(const v8h*)(p);
  u.h[1] = *(const v8h*)(p + 8);
  return u.v;
}

// B fragment from an LDS tile: B[k][n] = tile[(n0+n)*stride + col0 + k].
__device__ inline v16h frag_b_lds(const _Float16* tile, int stride, int n0, int col0, int lane) {
  const int kb = (lane & 16) ? 16 : 0;
  const _Float16* p = tile + (n0 + (lane & 15)) * stride + col0 + kb;
  union { v16h v; v8h h[2]; } u;
  u.h[0] = *(const v8h*)(p);
  u.h[1] = *(const v8h*)(p + 8);
  return u.v;
}

// ---------------------------------------------------------------------------
// Conversion kernels (fp32 inputs -> f16 workspace)
// ---------------------------------------------------------------------------

__global__ void cvt_f32_to_f16(const float* __restrict__ in, _Float16* __restrict__ out, int n) {
  int i = blockIdx.x * blockDim.x + threadIdx.x;
  if (i < n) out[i] = (_Float16)in[i];
}

// W[k][n] (K x N row-major) -> Wt[n][k] (N x K row-major) f16
__global__ void cvt_transpose_f16(const float* __restrict__ W, _Float16* __restrict__ Wt,
                                  int K, int N) {
  int i = blockIdx.x * blockDim.x + threadIdx.x;
  if (i < N * K) {
    int nrow = i / K;
    int k = i - nrow * K;
    Wt[i] = (_Float16)W[(size_t)k * N + nrow];
  }
}

// ---------------------------------------------------------------------------
// Copy-free software-pipelined WMMA GEMM: C[M,N] = A[M,K] @ Bt[N,K]^T + bias.
// 256 threads = 8 waves (2 M x 4 N); wave tile 32M x 64N (8 wmma / K-step).
// Two named fragment buffers alternate roles (distance-1 prefetch, no moves).
// ---------------------------------------------------------------------------

__global__ __launch_bounds__(256) void gemm_wmma_kernel(const _Float16* __restrict__ A,
                                                        const _Float16* __restrict__ Bt,
                                                        const float* __restrict__ bias,
                                                        _Float16* __restrict__ outh,
                                                        float* __restrict__ outf,
                                                        int K, int N) {
  const int lane = threadIdx.x & 31;
  const int w = threadIdx.x >> 5;
  const int wm = w & 1;
  const int wn = w >> 1;
  const size_t m0 = (size_t)blockIdx.y * 64 + wm * 32;
  const int n0 = blockIdx.x * 256 + wn * 64;

  v8f zero = {};
  v8f acc[2][4];
#pragma unroll
  for (int i = 0; i < 2; ++i)
#pragma unroll
    for (int j = 0; j < 4; ++j) acc[i][j] = zero;

  v16h a0[2], b0[4], a1[2], b1[4];

#define GEMM_LOAD(abuf, bbuf, kk)                                                   \
  do {                                                                              \
    _Pragma("unroll")                                                               \
    for (int i = 0; i < 2; ++i)                                                     \
      (abuf)[i] = frag_a_rowmajor(A, (size_t)K, m0 + i * 16, (kk), lane);           \
    _Pragma("unroll")                                                               \
    for (int j = 0; j < 4; ++j)                                                     \
      (bbuf)[j] = frag_b_colmajor(Bt, (size_t)K, (size_t)(n0 + j * 16), (kk), lane);\
  } while (0)

#define GEMM_MMA(abuf, bbuf)                                                        \
  do {                                                                              \
    _Pragma("unroll")                                                               \
    for (int i = 0; i < 2; ++i)                                                     \
      _Pragma("unroll")                                                             \
      for (int j = 0; j < 4; ++j)                                                   \
        acc[i][j] = __builtin_amdgcn_wmma_f32_16x16x32_f16(                         \
            false, (abuf)[i], false, (bbuf)[j], (short)0, acc[i][j], false, false); \
  } while (0)

  GEMM_LOAD(a0, b0, 0);
  GEMM_LOAD(a1, b1, 32);
  for (int k0 = 64; k0 <= K - 64; k0 += 64) {
    GEMM_MMA(a0, b0);
    GEMM_LOAD(a0, b0, k0);        // overlaps GEMM_MMA(a1, b1)
    GEMM_MMA(a1, b1);
    GEMM_LOAD(a1, b1, k0 + 32);   // overlaps next iteration's GEMM_MMA(a0, b0)
  }
  GEMM_MMA(a0, b0);
  GEMM_MMA(a1, b1);
#undef GEMM_LOAD
#undef GEMM_MMA

  const int mofs = (lane & 16) ? 8 : 0;   // C layout: VGPR r -> row r or r+8
  const int nl = lane & 15;
#pragma unroll
  for (int j = 0; j < 4; ++j) {
    const int col = n0 + j * 16 + nl;
    const float bv = bias[col];
#pragma unroll
    for (int i = 0; i < 2; ++i)
#pragma unroll
      for (int r = 0; r < 8; ++r) {
        const size_t row = m0 + i * 16 + (size_t)(r + mofs);
        const float v = acc[i][j][r] + bv;
        if (outh) outh[row * (size_t)N + col] = (_Float16)v;
        else      outf[row * (size_t)N + col] = v;
      }
  }
}

// ---------------------------------------------------------------------------
// Flash attention: grid (L/64, B*H), 128 threads = 4 waves; wave owns a
// 16-row query tile. Double-buffered LDS staging, pipelined one key-block
// ahead: K tiles via async TDM (TENSORcnt), V tiles via register prefetch.
// ---------------------------------------------------------------------------

__global__ __launch_bounds__(128) void flash_attn_kernel(const _Float16* __restrict__ qkvh,
                                                         _Float16* __restrict__ attnh) {
  __shared__ _Float16 kTile[2][32 * 64];    // [buf][key][d]
  __shared__ _Float16 vTile[2][64 * 40];    // [buf][d][key], stride 40 (bank-pad)
  __shared__ _Float16 plds[4][16 * 32];     // per-wave P staging

  const int lane = threadIdx.x & 31;
  const int w = threadIdx.x >> 5;
  const int b = blockIdx.y >> 4;            // / H
  const int h = blockIdx.y & 15;            // % H
  const int blockq0 = blockIdx.x * 64;
  const int q0 = blockq0 + w * 16;
  const size_t rowbase = (size_t)b * Lq;
  const size_t ld = (size_t)D3;
  const int mofs = (lane & 16) ? 8 : 0;
  const int nl = lane & 15;

  // Q fragments: two 16x32 A-frags covering d = 0..63
  const v16h qa0 = frag_a_rowmajor(qkvh + h * HD, ld, rowbase + q0, 0, lane);
  const v16h qa1 = frag_a_rowmajor(qkvh + h * HD, ld, rowbase + q0, 32, lane);

  float mrow[8], lrow[8];
  v8f zero = {};
  v8f acc[4] = {zero, zero, zero, zero};
#pragma unroll
  for (int r = 0; r < 8; ++r) { mrow[r] = -1e30f; lrow[r] = 0.0f; }

  const int my_nkb = (q0 + 15) / 32 + 1;        // this wave's causal limit
  const int nkb = (blockq0 + 63) / 32 + 1;      // block max (last wave)

  // cooperative staging indices: 128 threads cover 32 rows x 64 cols (16/thread)
  const int vr = threadIdx.x >> 2;              // key row 0..31
  const int vc = (threadIdx.x & 3) * 16;        // d col {0,16,32,48}

#ifdef USE_TDM
  // Issue an async TDM load of the 32x64 f16 K tile for key-block kb into buf.
  auto tdm_k = [&](int kb, int buf) {
    const _Float16* gk = qkvh + (rowbase + kb * 32) * ld + Dm + h * HD;
    const unsigned long long ga = (unsigned long long)(size_t)gk;
    const unsigned ldsOff = (unsigned)(size_t)(&kTile[buf][0]);
    // D# group0: count=1 | lds_addr | global_addr | type=2 ("image")
    v4u g0 = { 1u, ldsOff, (unsigned)ga,
               (unsigned)((ga >> 32) & 0x1FFFFFFu) | (2u << 30) };
    // D# group1: data_size=2B; tensor 64x32; tile 64x32; dim0 stride=3072 elts
    v8i_ g1 = { (int)(1u << 16),        // [17:16] data_size = 1 (2 bytes)
                (int)(64u << 16),       // tensor_dim0[15:0] @ bits 63:48
                (int)(32u << 16),       // tensor_dim0 hi=0 | tensor_dim1[15:0]
                (int)(64u << 16),       // tensor_dim1 hi=0 | tile_dim0=64
                (int)32,                // tile_dim1=32 | tile_dim2=0
                (int)3072,              // tensor_dim0_stride[31:0]
                0, 0 };
    v4i_ gz4 = { 0, 0, 0, 0 };
    v8i_ gz8 = { 0, 0, 0, 0, 0, 0, 0, 0 };
    __builtin_amdgcn_tensor_load_to_lds(g0, g1, gz4, gz4, gz8, 0);
  };
#endif

  // ---- prologue: start staging key-block 0 ----
  v8h vx0, vx1;                 // V prefetch registers (this thread's 16 halves)
#ifndef USE_TDM
  v8h kx0, kx1;                 // K prefetch registers (fallback path)
#endif
  {
    const _Float16* vsrc = qkvh + (rowbase + vr) * ld + 2 * Dm + h * HD + vc;
    vx0 = *(const v8h*)vsrc;
    vx1 = *(const v8h*)(vsrc + 8);
#ifdef USE_TDM
    if (w == 0) tdm_k(0, 0);
#else
    const _Float16* ksrc = qkvh + (rowbase + vr) * ld + Dm + h * HD + vc;
    kx0 = *(const v8h*)ksrc;
    kx1 = *(const v8h*)(ksrc + 8);
#endif
  }

  for (int kb = 0; kb < nkb; ++kb) {
    const int buf = kb & 1;
    const int kbase = kb * 32;

    // ---- commit staged data for kb into LDS buffer `buf` ----
    {
      _Float16* vt = vTile[buf];
#pragma unroll
      for (int e = 0; e < 8; ++e) {
        vt[(vc + e) * 40 + vr]     = vx0[e];
        vt[(vc + 8 + e) * 40 + vr] = vx1[e];
      }
#ifndef USE_TDM
      _Float16* kt = kTile[buf];
      *(v8h*)(kt + vr * 64 + vc)     = kx0;
      *(v8h*)(kt + vr * 64 + vc + 8) = kx1;
#endif
    }
#ifdef USE_TDM
    if (w == 0) __builtin_amdgcn_s_wait_tensorcnt((short)0);  // K tile kb done
#endif
    __syncthreads();   // kTile[buf], vTile[buf] visible to all waves

    // ---- start staging kb+1 (overlaps the compute below) ----
    if (kb + 1 < nkb) {
      const _Float16* vsrc = qkvh + (rowbase + kbase + 32 + vr) * ld + 2 * Dm + h * HD + vc;
      vx0 = *(const v8h*)vsrc;
      vx1 = *(const v8h*)(vsrc + 8);
#ifdef USE_TDM
      if (w == 0) tdm_k(kb + 1, buf ^ 1);    // async; waited next iteration
#else
      const _Float16* ksrc = qkvh + (rowbase + kbase + 32 + vr) * ld + Dm + h * HD + vc;
      kx0 = *(const v8h*)ksrc;
      kx1 = *(const v8h*)(ksrc + 8);
#endif
    }

    if (kb < my_nkb) {
      const _Float16* kt = kTile[buf];
      const _Float16* vt = vTile[buf];

      // ---- S = Q @ K^T (two 16x16 tiles over 32 keys) ----
      v8f s[2];
#pragma unroll
      for (int t = 0; t < 2; ++t) {
        v8f st = {};
        v16h kf0 = frag_b_lds(kt, 64, t * 16, 0, lane);    // d = 0..31
        st = __builtin_amdgcn_wmma_f32_16x16x32_f16(false, qa0, false, kf0, (short)0, st, false, false);
        v16h kf1 = frag_b_lds(kt, 64, t * 16, 32, lane);   // d = 32..63
        st = __builtin_amdgcn_wmma_f32_16x16x32_f16(false, qa1, false, kf1, (short)0, st, false, false);
        s[t] = st;
      }

      // ---- scale + causal mask ----
#pragma unroll
      for (int t = 0; t < 2; ++t) {
        const int ki = kbase + t * 16 + nl;
#pragma unroll
        for (int r = 0; r < 8; ++r) {
          const int qi = q0 + r + mofs;
          const float sv = s[t][r] * 0.125f;      // HD^-0.5
          s[t][r] = (ki <= qi) ? sv : -1e30f;
        }
      }

      // ---- online softmax (row stats across 16-lane half-groups) ----
      float cf[8];
#pragma unroll
      for (int r = 0; r < 8; ++r) {
        float bm = fmaxf(s[0][r], s[1][r]);
#pragma unroll
        for (int off = 8; off >= 1; off >>= 1)
          bm = fmaxf(bm, __shfl_xor(bm, off, 32));
        const float mnew = fmaxf(mrow[r], bm);
        cf[r] = __expf(mrow[r] - mnew);
        mrow[r] = mnew;
        const float p0 = __expf(s[0][r] - mnew);
        const float p1 = __expf(s[1][r] - mnew);
        s[0][r] = p0;
        s[1][r] = p1;
        float rs = p0 + p1;
#pragma unroll
        for (int off = 8; off >= 1; off >>= 1)
          rs += __shfl_xor(rs, off, 32);
        lrow[r] = lrow[r] * cf[r] + rs;
      }
#pragma unroll
      for (int dt = 0; dt < 4; ++dt)
#pragma unroll
        for (int r = 0; r < 8; ++r) acc[dt][r] *= cf[r];

      // ---- P: C-layout f32 -> A-layout f16 via per-wave LDS ----
      _Float16* pl = plds[w];
#pragma unroll
      for (int t = 0; t < 2; ++t)
#pragma unroll
        for (int r = 0; r < 8; ++r)
          pl[(r + mofs) * 32 + t * 16 + nl] = (_Float16)s[t][r];

      asm volatile("s_wait_dscnt 0" ::: "memory");

      v16h pa;
      {
        const int kb8 = (lane & 16) ? 16 : 0;
        union { v16h v; v8h hh[2]; } u;
        u.hh[0] = *(const v8h*)(pl + nl * 32 + kb8);
        u.hh[1] = *(const v8h*)(pl + nl * 32 + kb8 + 8);
        pa = u.v;
      }

      // ---- O += P @ V (4 n-tiles over HD=64, B-frags from transposed vTile) ----
#pragma unroll
      for (int dt = 0; dt < 4; ++dt) {
        v16h vf = frag_b_lds(vt, 40, dt * 16, 0, lane);
        acc[dt] = __builtin_amdgcn_wmma_f32_16x16x32_f16(
            false, pa, false, vf, (short)0, acc[dt], false, false);
      }
    }
    // No trailing barrier needed: buffer `buf` is only rewritten after the barrier of
    // iteration kb+1 (TDM K load issued in kb+1; V and fallback K at kb+2's commit).
  }

  // ---- epilogue: O /= rowsum, store f16 [BL, D] ----
#pragma unroll
  for (int r = 0; r < 8; ++r) {
    const float inv = 1.0f / lrow[r];
    const size_t row = rowbase + q0 + (size_t)(r + mofs);
#pragma unroll
    for (int dt = 0; dt < 4; ++dt)
      attnh[row * (size_t)Dm + h * HD + dt * 16 + nl] = (_Float16)(acc[dt][r] * inv);
  }
}

// ---------------------------------------------------------------------------
// Launch: cvt x3 -> QKV GEMM -> flash attention -> proj GEMM
// ---------------------------------------------------------------------------

extern "C" void kernel_launch(void* const* d_in, const int* in_sizes, int n_in,
                              void* d_out, int out_size, void* d_ws, size_t ws_size,
                              hipStream_t stream) {
  const float* x     = (const float*)d_in[0];
  const float* Wqkv  = (const float*)d_in[1];
  const float* bqkv  = (const float*)d_in[2];
  const float* Wproj = (const float*)d_in[3];
  const float* bproj = (const float*)d_in[4];
  float* out = (float*)d_out;

  // workspace carve (f16): xh | Wqkv^T | Wproj^T | qkv | attn_out  (~48 MB)
  _Float16* xh     = (_Float16*)d_ws;
  _Float16* wqkvt  = xh     + (size_t)BL * Dm;
  _Float16* wprojt = wqkvt  + (size_t)D3 * Dm;
  _Float16* qkvh   = wprojt + (size_t)Dm * Dm;
  _Float16* attnh  = qkvh   + (size_t)BL * D3;

  {
    int n = BL * Dm;
    cvt_f32_to_f16<<<(n + 255) / 256, 256, 0, stream>>>(x, xh, n);
  }
  {
    int n = D3 * Dm;
    cvt_transpose_f16<<<(n + 255) / 256, 256, 0, stream>>>(Wqkv, wqkvt, Dm, D3);
  }
  {
    int n = Dm * Dm;
    cvt_transpose_f16<<<(n + 255) / 256, 256, 0, stream>>>(Wproj, wprojt, Dm, Dm);
  }

  // QKV: [4096,1024] @ [1024,3072] + bqkv -> f16 [4096,3072]
  gemm_wmma_kernel<<<dim3(D3 / 256, BL / 64), 256, 0, stream>>>(
      xh, wqkvt, bqkv, qkvh, nullptr, Dm, D3);

  // Flash attention -> f16 [4096,1024]
  flash_attn_kernel<<<dim3(Lq / 64, Bb * Hh), 128, 0, stream>>>(qkvh, attnh);

  // Proj: [4096,1024] @ [1024,1024] + bproj -> f32 output
  gemm_wmma_kernel<<<dim3(Dm / 256, BL / 64), 256, 0, stream>>>(
      attnh, wprojt, bproj, nullptr, out, Dm, Dm);
}
